// bertCRF_34969623724375
// MI455X (gfx1250) — matrix operand order repaired and checked by reading, not compiled, not run on hardware
//
#include <hip/hip_runtime.h>
#include <hip/hip_bf16.h>
#include <math.h>

typedef __attribute__((ext_vector_type(16))) __bf16 v16bf;
typedef __attribute__((ext_vector_type(8)))  __bf16 v8bf;
typedef __attribute__((ext_vector_type(8)))  float  v8f;

#define BB   8
#define SS   512
#define HH   768
#define LLAY 12
#define NHH  12
#define DHH  64
#define FFF  3072
#define CC   9
#define MM   (BB*SS)   /* 4096 rows */

static __device__ __forceinline__ v8f wmma_bf16(v16bf a, v16bf b, v8f c) {
  // D = A(16x32 bf16) * B(32x16 bf16) + C(16x16 f32)
  return __builtin_amdgcn_wmma_f32_16x16x32_bf16(false, a, false, b, (short)0, c, false, false);
}
static __device__ __forceinline__ v16bf cat8(v8bf lo, v8bf hi) {
  return __builtin_shufflevector(lo, hi, 0,1,2,3,4,5,6,7,8,9,10,11,12,13,14,15);
}
static __device__ __forceinline__ v8bf cvt8(float4 f0, float4 f1) {
  v8bf r;
  r[0]=(__bf16)f0.x; r[1]=(__bf16)f0.y; r[2]=(__bf16)f0.z; r[3]=(__bf16)f0.w;
  r[4]=(__bf16)f1.x; r[5]=(__bf16)f1.y; r[6]=(__bf16)f1.z; r[7]=(__bf16)f1.w;
  return r;
}

// ---------------------------------------------------------------------------
// Tiled WMMA GEMM: C[M,N] = act( A[M,K] * B[K,N] + bias[N] )
// BM=128, BN=64, BK=32. 256 threads = 8 waves; each wave owns a 32x32 block
// (4 accumulators, 4 WMMA per k-step). Register-buffered software pipeline:
// next tile's global loads are issued before the WMMA section; the tile two
// steps ahead is warmed in GL2 with global_prefetch_b8.
// ---------------------------------------------------------------------------
template<int ACT>
__global__ __launch_bounds__(256) void gemm_bias_kernel(
    const float* __restrict__ A, const float* __restrict__ Bw,
    const float* __restrict__ bias, float* __restrict__ C,
    int M, int N, int K)
{
  __shared__ alignas(16) __bf16 As[128][48];   // [m][k]
  __shared__ alignas(16) __bf16 BsT[64][48];   // [n][k] (transposed)
  const int t  = threadIdx.x;
  const int l  = t & 31, w = t >> 5;
  const int wy = w & 3,  wx = w >> 2;
  const int hi = l >> 4, ln = l & 15;
  const int m0 = blockIdx.y * 128, n0 = blockIdx.x * 64;
  const int ar = t >> 1, ac = (t & 1) * 16;          // A stage: 16 elems/thread
  const int bn = t & 63, bk = (t >> 6) * 8;          // B stage: 8 elems/thread
  v8f acc00 = {}, acc01 = {}, acc10 = {}, acc11 = {};

  const float* aRow = A  + (size_t)(m0 + ar) * K + ac;
  const float* bCol = Bw + (size_t)bk * N + n0 + bn;

  // --- pipeline prologue: load tile k0=0 into registers ---
  float4 fa0, fa1, fa2, fa3;
  float  fb[8];
  {
    const float4* sa = (const float4*)aRow;
    fa0 = sa[0]; fa1 = sa[1]; fa2 = sa[2]; fa3 = sa[3];
#pragma unroll
    for (int j = 0; j < 8; ++j) fb[j] = bCol[(size_t)j * N];
  }

  for (int k0 = 0; k0 < K; k0 += 32) {
    // store register-staged tile into LDS (packed bf16, b128 stores)
    *(v8bf*)&As[ar][ac]     = cvt8(fa0, fa1);
    *(v8bf*)&As[ar][ac + 8] = cvt8(fa2, fa3);
    {
      v8bf bb;
#pragma unroll
      for (int j = 0; j < 8; ++j) bb[j] = (__bf16)fb[j];
      *(v8bf*)&BsT[bn][bk] = bb;
    }
    __syncthreads();

    // issue next tile's global loads early (overlaps frag loads + WMMA)
    if (k0 + 32 < K) {
      const float4* sa = (const float4*)(aRow + k0 + 32);
      fa0 = sa[0]; fa1 = sa[1]; fa2 = sa[2]; fa3 = sa[3];
      const float* bc = bCol + (size_t)(k0 + 32) * N;
#pragma unroll
      for (int j = 0; j < 8; ++j) fb[j] = bc[(size_t)j * N];
    }
    if (k0 + 64 < K) {  // warm GL2 two tiles ahead -> global_prefetch_b8
      __builtin_prefetch(aRow + k0 + 64, 0, 1);
      __builtin_prefetch(bCol + (size_t)(k0 + 64) * N, 0, 1);
    }

    const int mA = wy * 32 + ln, mB = mA + 16;
    const v16bf a0 = cat8(*(const v8bf*)&As[mA][hi * 8],
                          *(const v8bf*)&As[mA][16 + hi * 8]);
    const v16bf a1 = cat8(*(const v8bf*)&As[mB][hi * 8],
                          *(const v8bf*)&As[mB][16 + hi * 8]);
    const int nA = wx * 32 + ln, nB = nA + 16;
    const v16bf b0 = cat8(*(const v8bf*)&BsT[nA][hi * 8],
                          *(const v8bf*)&BsT[nA][16 + hi * 8]);
    const v16bf b1 = cat8(*(const v8bf*)&BsT[nB][hi * 8],
                          *(const v8bf*)&BsT[nB][16 + hi * 8]);
    acc00 = wmma_bf16(a0, b0, acc00);
    acc01 = wmma_bf16(a0, b1, acc01);
    acc10 = wmma_bf16(a1, b0, acc10);
    acc11 = wmma_bf16(a1, b1, acc11);
    __syncthreads();
  }

  const int nA = n0 + wx * 32 + ln, nB = nA + 16;
  const float bA = bias[nA], bBv = bias[nB];
#pragma unroll
  for (int i = 0; i < 8; ++i) {
    const int mg0 = m0 + wy * 32 + i + 8 * hi;
    const int mg1 = mg0 + 16;
    float v00 = acc00[i] + bA, v01 = acc01[i] + bBv;
    float v10 = acc10[i] + bA, v11 = acc11[i] + bBv;
    if (ACT) {  // exact GELU
      v00 = 0.5f * v00 * (1.0f + erff(v00 * 0.70710678118654752f));
      v01 = 0.5f * v01 * (1.0f + erff(v01 * 0.70710678118654752f));
      v10 = 0.5f * v10 * (1.0f + erff(v10 * 0.70710678118654752f));
      v11 = 0.5f * v11 * (1.0f + erff(v11 * 0.70710678118654752f));
    }
    C[(size_t)mg0 * N + nA] = v00;
    C[(size_t)mg0 * N + nB] = v01;
    C[(size_t)mg1 * N + nA] = v10;
    C[(size_t)mg1 * N + nB] = v11;
  }
}

// ---------------------------------------------------------------------------
// Fused attention: one block per (b, h, 64-query tile). Score panel (f32) and
// probability panel (bf16) live in LDS (~213KB of CDNA5's 320KB WGP LDS).
// QK^T and P*V via WMMA; fragments are all b128 LDS loads.
// ---------------------------------------------------------------------------
__global__ __launch_bounds__(256) void attention_kernel(
    const float* __restrict__ qkv, const int* __restrict__ amask,
    float* __restrict__ ctx)
{
  extern __shared__ char smem[];
  __bf16* Qs  = (__bf16*)smem;                          // 64x72 bf16 [q][d]
  __bf16* KVs = Qs + 64 * 72;                           // 64x72 bf16 K:[n][d], V:[d][k]
  float*  sc  = (float*)(smem + 2 * 64 * 72 * 2);       // 64x520 f32 scores
  __bf16* pb  = (__bf16*)(sc + 64 * 520);               // 64x520 bf16 probs

  const int t  = threadIdx.x, l = t & 31, w = t >> 5;
  const int wy = w & 3, wx = w >> 2, hi = l >> 4, ln = l & 15;
  const int qt = blockIdx.x, h = blockIdx.y, b = blockIdx.z;
  const int q0 = qt * 64;
  const size_t rs = 3 * HH;  // qkv row stride = 2304

  // Stage Q (pre-scaled by 1/sqrt(DH)=0.125), 16 elems/thread, b128 stores
  {
    const int r = t >> 2, c0 = (t & 3) * 16;
    const float4* src = (const float4*)(qkv + (size_t)(b * SS + q0 + r) * rs + h * DHH + c0);
    float4 f0 = src[0], f1 = src[1], f2 = src[2], f3 = src[3];
    f0.x*=0.125f; f0.y*=0.125f; f0.z*=0.125f; f0.w*=0.125f;
    f1.x*=0.125f; f1.y*=0.125f; f1.z*=0.125f; f1.w*=0.125f;
    f2.x*=0.125f; f2.y*=0.125f; f2.z*=0.125f; f2.w*=0.125f;
    f3.x*=0.125f; f3.y*=0.125f; f3.z*=0.125f; f3.w*=0.125f;
    *(v8bf*)&Qs[r * 72 + c0]     = cvt8(f0, f1);
    *(v8bf*)&Qs[r * 72 + c0 + 8] = cvt8(f2, f3);
  }
  __syncthreads();

  // ---- Phase 1: S = Q K^T + mask bias ----
  const int m = wy * 16 + ln;
  for (int kt = 0; kt < 8; ++kt) {
    {
      const int r = t >> 2, c0 = (t & 3) * 16;  // K tile [key n][d]
      const float* base = qkv + (size_t)(b * SS + kt * 64 + r) * rs + HH + h * DHH + c0;
      const float4* src = (const float4*)base;
      float4 f0 = src[0], f1 = src[1], f2 = src[2], f3 = src[3];
      *(v8bf*)&KVs[r * 72 + c0]     = cvt8(f0, f1);
      *(v8bf*)&KVs[r * 72 + c0 + 8] = cvt8(f2, f3);
      if (kt < 7)  // warm GL2 for the next K tile
        __builtin_prefetch(base + 64 * rs, 0, 1);
    }
    __syncthreads();

    const v16bf a0 = cat8(*(const v8bf*)&Qs[m * 72 + hi * 8],
                          *(const v8bf*)&Qs[m * 72 + 16 + hi * 8]);
    const v16bf a1 = cat8(*(const v8bf*)&Qs[m * 72 + 32 + hi * 8],
                          *(const v8bf*)&Qs[m * 72 + 48 + hi * 8]);
#pragma unroll
    for (int tc = 0; tc < 2; ++tc) {
      const int n = wx * 32 + tc * 16 + ln;     // local key index = B column
      const v16bf b0 = cat8(*(const v8bf*)&KVs[n * 72 + hi * 8],
                            *(const v8bf*)&KVs[n * 72 + 16 + hi * 8]);
      const v16bf b1 = cat8(*(const v8bf*)&KVs[n * 72 + 32 + hi * 8],
                            *(const v8bf*)&KVs[n * 72 + 48 + hi * 8]);
      v8f acc = {};
      acc = wmma_bf16(a0, b0, acc);
      acc = wmma_bf16(a1, b1, acc);
      const int key = kt * 64 + n;
      const float mb = (1.0f - (float)amask[b * SS + key]) * -10000.0f;
#pragma unroll
      for (int i = 0; i < 8; ++i)
        sc[(wy * 16 + i + 8 * hi) * 520 + key] = acc[i] + mb;
    }
    __syncthreads();
  }

  // ---- Phase 2: softmax rows -> bf16 prob panel (each wave: 8 rows) ----
  for (int q = 0; q < 8; ++q) {
    const int r = w * 8 + q;
    const float* srow = sc + r * 520 + l * 16;   // 16 contiguous elems/lane
    float e[16];
    float mx = -1e30f;
#pragma unroll
    for (int j = 0; j < 16; ++j) { e[j] = srow[j]; mx = fmaxf(mx, e[j]); }
#pragma unroll
    for (int o = 16; o > 0; o >>= 1) mx = fmaxf(mx, __shfl_xor(mx, o, 32));
    float sum = 0.f;
#pragma unroll
    for (int j = 0; j < 16; ++j) { e[j] = __expf(e[j] - mx); sum += e[j]; }
#pragma unroll
    for (int o = 16; o > 0; o >>= 1) sum += __shfl_xor(sum, o, 32);
    const float inv = 1.0f / sum;
    v8bf p0, p1;
#pragma unroll
    for (int j = 0; j < 8; ++j) { p0[j] = (__bf16)(e[j] * inv); p1[j] = (__bf16)(e[8 + j] * inv); }
    *(v8bf*)&pb[r * 520 + l * 16]     = p0;
    *(v8bf*)&pb[r * 520 + l * 16 + 8] = p1;
  }
  __syncthreads();

  // ---- Phase 3: ctx = P V (V staged transposed [d][k]) ----
  v8f acc0 = {}, acc1 = {};
  for (int kt = 0; kt < 8; ++kt) {
    {
      const int d = t & 63, kb = (t >> 6) * 16;
      const float* base = qkv + (size_t)(b * SS + kt * 64 + kb) * rs + 2 * HH + h * DHH + d;
      v8bf v0, v1;
#pragma unroll
      for (int j = 0; j < 8; ++j) {
        v0[j] = (__bf16)base[(size_t)j * rs];
        v1[j] = (__bf16)base[(size_t)(8 + j) * rs];
      }
      *(v8bf*)&KVs[d * 72 + kb]     = v0;
      *(v8bf*)&KVs[d * 72 + kb + 8] = v1;
      if (kt < 7)  // warm GL2 for the next V tile
        __builtin_prefetch(base + 64 * rs, 0, 1);
    }
    __syncthreads();
#pragma unroll
    for (int kk = 0; kk < 2; ++kk) {
      const int ks = kt * 64 + kk * 32;
      const v16bf a = cat8(*(const v8bf*)&pb[m * 520 + ks + hi * 8],
                           *(const v8bf*)&pb[m * 520 + ks + 16 + hi * 8]);
      const int dA = wx * 32 + ln, dB = dA + 16;
      const v16bf b0 = cat8(*(const v8bf*)&KVs[dA * 72 + kk * 32 + hi * 8],
                            *(const v8bf*)&KVs[dA * 72 + kk * 32 + 16 + hi * 8]);
      const v16bf b1 = cat8(*(const v8bf*)&KVs[dB * 72 + kk * 32 + hi * 8],
                            *(const v8bf*)&KVs[dB * 72 + kk * 32 + 16 + hi * 8]);
      acc0 = wmma_bf16(a, b0, acc0);
      acc1 = wmma_bf16(a, b1, acc1);
    }
    __syncthreads();
  }
  const int dA = wx * 32 + ln, dB = dA + 16;
#pragma unroll
  for (int i = 0; i < 8; ++i) {
    const int ml = wy * 16 + i + 8 * hi;
    const size_t row = (size_t)(b * SS + q0 + ml);
    ctx[row * HH + h * DHH + dA] = acc0[i];
    ctx[row * HH + h * DHH + dB] = acc1[i];
  }
}

// ---------------------------------------------------------------------------
// Embedding gather + LayerNorm (1 block per token row)
// ---------------------------------------------------------------------------
__global__ __launch_bounds__(256) void embed_ln_kernel(
    const int* __restrict__ ids, const float* __restrict__ tok,
    const float* __restrict__ pos, const float* __restrict__ typ,
    const float* __restrict__ g, const float* __restrict__ bt,
    float* __restrict__ x)
{
  __shared__ float red[256];
  const int row = blockIdx.x, t = threadIdx.x;
  const int s = row & (SS - 1);
  const int id = ids[row];
  float v[3];
#pragma unroll
  for (int j = 0; j < 3; ++j) {
    const int c = t + j * 256;
    v[j] = tok[(size_t)id * HH + c] + pos[(size_t)s * HH + c] + typ[c];
  }
  red[t] = v[0] + v[1] + v[2];
  __syncthreads();
  for (int o = 128; o > 0; o >>= 1) { if (t < o) red[t] += red[t + o]; __syncthreads(); }
  const float mean = red[0] / (float)HH;
  __syncthreads();
  float sq = 0.f;
#pragma unroll
  for (int j = 0; j < 3; ++j) { const float d = v[j] - mean; sq += d * d; }
  red[t] = sq;
  __syncthreads();
  for (int o = 128; o > 0; o >>= 1) { if (t < o) red[t] += red[t + o]; __syncthreads(); }
  const float rstd = rsqrtf(red[0] / (float)HH + 1e-12f);
#pragma unroll
  for (int j = 0; j < 3; ++j) {
    const int c = t + j * 256;
    x[(size_t)row * HH + c] = (v[j] - mean) * rstd * g[c] + bt[c];
  }
}

// ---------------------------------------------------------------------------
// x = LN(x + delta) (residual; GEMM bias already folded into delta)
// ---------------------------------------------------------------------------
__global__ __launch_bounds__(256) void ln_residual_kernel(
    const float* __restrict__ xin, const float* __restrict__ delta,
    const float* __restrict__ g, const float* __restrict__ bt,
    float* __restrict__ xout)
{
  __shared__ float red[256];
  const int row = blockIdx.x, t = threadIdx.x;
  float v[3];
#pragma unroll
  for (int j = 0; j < 3; ++j) {
    const int c = t + j * 256;
    v[j] = xin[(size_t)row * HH + c] + delta[(size_t)row * HH + c];
  }
  red[t] = v[0] + v[1] + v[2];
  __syncthreads();
  for (int o = 128; o > 0; o >>= 1) { if (t < o) red[t] += red[t + o]; __syncthreads(); }
  const float mean = red[0] / (float)HH;
  __syncthreads();
  float sq = 0.f;
#pragma unroll
  for (int j = 0; j < 3; ++j) { const float d = v[j] - mean; sq += d * d; }
  red[t] = sq;
  __syncthreads();
  for (int o = 128; o > 0; o >>= 1) { if (t < o) red[t] += red[t + o]; __syncthreads(); }
  const float rstd = rsqrtf(red[0] / (float)HH + 1e-12f);
#pragma unroll
  for (int j = 0; j < 3; ++j) {
    const int c = t + j * 256;
    xout[(size_t)row * HH + c] = (v[j] - mean) * rstd * g[c] + bt[c];
  }
}

// ---------------------------------------------------------------------------
// Head: em[row, 0..8] = x[row,:] . head_w[:,c] + head_b[c]  (wave per class)
// ---------------------------------------------------------------------------
__global__ __launch_bounds__(256) void head_kernel(
    const float* __restrict__ x, const float* __restrict__ hw,
    const float* __restrict__ hb, float* __restrict__ em)
{
  __shared__ float xs[HH];
  const int row = blockIdx.x, t = threadIdx.x;
  const int l = t & 31, wv = t >> 5;
#pragma unroll
  for (int j = 0; j < 3; ++j) xs[t + j * 256] = x[(size_t)row * HH + t + j * 256];
  __syncthreads();
  for (int c = wv; c < CC; c += 8) {
    float s = 0.f;
    for (int k = l; k < HH; k += 32) s += xs[k] * hw[(size_t)k * CC + c];
#pragma unroll
    for (int o = 16; o > 0; o >>= 1) s += __shfl_xor(s, o, 32);
    if (l == 0) em[(size_t)row * CC + c] = s + hb[c];
  }
}

__global__ void init_out_kernel(float* o) { o[0] = 0.f; }

// ---------------------------------------------------------------------------
// CRF log-likelihood: one wave per batch item, alpha in lanes 0..8
// ---------------------------------------------------------------------------
__global__ __launch_bounds__(32) void crf_forward_kernel(
    const float* __restrict__ em, const int* __restrict__ labels,
    const int* __restrict__ mask, const float* __restrict__ trans,
    const float* __restrict__ st, const float* __restrict__ et,
    float* __restrict__ out)
{
  __shared__ float tr[CC * CC];
  const int b = blockIdx.x, l = threadIdx.x;
  for (int i = l; i < CC * CC; i += 32) tr[i] = trans[i];
  __syncthreads();
  const float NEG = -1e30f;
  float alpha = (l < CC) ? st[l] + em[(size_t)(b * SS) * CC + l] : NEG;

  float num = 0.f; int prevtag = 0, lasttag = 0;
  if (l == 0) {
    prevtag = labels[b * SS];
    lasttag = prevtag;
    num = st[prevtag] + em[(size_t)(b * SS) * CC + prevtag];
  }
  const int jj = (l < CC) ? l : 0;
  for (int t = 1; t < SS; ++t) {
    const int mt = mask[b * SS + t];
    const float e = (l < CC) ? em[(size_t)(b * SS + t) * CC + l] : 0.f;
    float vs[CC], mx = NEG;
#pragma unroll
    for (int i = 0; i < CC; ++i) {
      const float ai = __shfl(alpha, i, 32);
      vs[i] = ai + tr[i * CC + jj];
      mx = fmaxf(mx, vs[i]);
    }
    float sum = 0.f;
#pragma unroll
    for (int i = 0; i < CC; ++i) sum += __expf(vs[i] - mx);
    const float nxt = mx + __logf(sum) + e;
    if (l < CC && mt) alpha = nxt;
    if (l == 0) {
      const int tg = labels[b * SS + t];
      if (mt) {
        num += tr[prevtag * CC + tg] + em[(size_t)(b * SS + t) * CC + tg];
        prevtag = tg; lasttag = tg;
      }
    }
  }
  if (l == 0) num += et[lasttag];

  float v = (l < CC) ? alpha + et[l] : NEG;
  float mx = v;
#pragma unroll
  for (int o = 16; o > 0; o >>= 1) mx = fmaxf(mx, __shfl_xor(mx, o, 32));
  float sum = __expf(v - mx);
#pragma unroll
  for (int o = 16; o > 0; o >>= 1) sum += __shfl_xor(sum, o, 32);
  const float logZ = mx + __logf(sum);
  if (l == 0) atomicAdd(out, -(num - logZ));
}

// ---------------------------------------------------------------------------
// Viterbi: forward max-product per wave, backtrack on lane 0.
// ---------------------------------------------------------------------------
__global__ __launch_bounds__(32) void viterbi_kernel(
    const float* __restrict__ em, const int* __restrict__ mask,
    const float* __restrict__ trans, const float* __restrict__ st,
    const float* __restrict__ et, int* __restrict__ hist,
    float* __restrict__ tags_out)
{
  __shared__ float tr[CC * CC];
  __shared__ float fin[CC];
  const int b = blockIdx.x, l = threadIdx.x;
  for (int i = l; i < CC * CC; i += 32) tr[i] = trans[i];
  __syncthreads();
  const float NEG = -1e30f;
  float score = (l < CC) ? st[l] + em[(size_t)(b * SS) * CC + l] : NEG;
  const int jj = (l < CC) ? l : 0;
  for (int t = 1; t < SS; ++t) {
    const int mt = mask[b * SS + t];
    const float e = (l < CC) ? em[(size_t)(b * SS + t) * CC + l] : 0.f;
    float best = NEG; int bi = 0;
#pragma unroll
    for (int i = 0; i < CC; ++i) {
      const float ai = __shfl(score, i, 32);
      const float vv = ai + tr[i * CC + jj];
      if (vv > best) { best = vv; bi = i; }
    }
    if (l < CC) {
      hist[((size_t)b * (SS - 1) + (t - 1)) * CC + l] = bi;
      if (mt) score = best + e;
    }
  }
  if (l < CC) fin[l] = score + et[l];
  __syncthreads();
  if (l == 0) {
    int cur = 0; float bv = NEG;
    for (int j = 0; j < CC; ++j) if (fin[j] > bv) { bv = fin[j]; cur = j; }
    for (int t = SS - 1; t >= 1; --t) {
      tags_out[b * SS + t] = (float)cur;
      const int prev = hist[((size_t)b * (SS - 1) + (t - 1)) * CC + cur];
      if (mask[b * SS + t]) cur = prev;
    }
    tags_out[b * SS] = (float)cur;
  }
}

// ---------------------------------------------------------------------------
extern "C" void kernel_launch(void* const* d_in, const int* in_sizes, int n_in,
                              void* d_out, int out_size, void* d_ws, size_t ws_size,
                              hipStream_t stream)
{
  (void)in_sizes; (void)n_in; (void)out_size; (void)ws_size;
  const int*   input_ids = (const int*)d_in[0];
  const int*   amask     = (const int*)d_in[1];
  const int*   labels    = (const int*)d_in[2];
  const float* tok_emb   = (const float*)d_in[3];
  const float* pos_emb   = (const float*)d_in[4];
  const float* type_emb  = (const float*)d_in[5];
  const float* eln_s     = (const float*)d_in[6];
  const float* eln_b     = (const float*)d_in[7];
  const float* qkv_w     = (const float*)d_in[8];
  const float* qkv_b     = (const float*)d_in[9];
  const float* ao_w      = (const float*)d_in[10];
  const float* ao_b      = (const float*)d_in[11];
  const float* ln1_s     = (const float*)d_in[12];
  const float* ln1_b     = (const float*)d_in[13];
  const float* ff1_w     = (const float*)d_in[14];
  const float* ff1_b     = (const float*)d_in[15];
  const float* ff2_w     = (const float*)d_in[16];
  const float* ff2_b     = (const float*)d_in[17];
  const float* ln2_s     = (const float*)d_in[18];
  const float* ln2_b     = (const float*)d_in[19];
  const float* head_w    = (const float*)d_in[20];
  const float* head_b    = (const float*)d_in[21];
  const float* trans     = (const float*)d_in[22];
  const float* start_t   = (const float*)d_in[23];
  const float* end_t     = (const float*)d_in[24];
  float* out = (float*)d_out;

  char* ws = (char*)d_ws;
  size_t off = 0;
  float* x     = (float*)(ws + off); off += (size_t)MM * HH * 4;
  float* qkv   = (float*)(ws + off); off += (size_t)MM * 3 * HH * 4;
  float* ctx   = (float*)(ws + off); off += (size_t)MM * HH * 4;
  float* delta = (float*)(ws + off); off += (size_t)MM * HH * 4;
  float* hmid  = (float*)(ws + off); off += (size_t)MM * FFF * 4;
  float* em    = (float*)(ws + off); off += (size_t)MM * CC * 4;
  int*   hist  = (int*)(ws + off);

  init_out_kernel<<<1, 1, 0, stream>>>(out);
  embed_ln_kernel<<<MM, 256, 0, stream>>>(input_ids, tok_emb, pos_emb, type_emb,
                                          eln_s, eln_b, x);

  const size_t attn_smem = (size_t)2 * 64 * 72 * 2      // Qs + KVs (bf16)
                         + (size_t)64 * 520 * 4         // scores (f32)
                         + (size_t)64 * 520 * 2;        // probs (bf16)
  for (int l = 0; l < LLAY; ++l) {
    gemm_bias_kernel<0><<<dim3(3 * HH / 64, MM / 128), 256, 0, stream>>>(
        x, qkv_w + (size_t)l * HH * 3 * HH, qkv_b + (size_t)l * 3 * HH,
        qkv, MM, 3 * HH, HH);
    attention_kernel<<<dim3(SS / 64, NHH, BB), 256, attn_smem, stream>>>(
        qkv, amask, ctx);
    gemm_bias_kernel<0><<<dim3(HH / 64, MM / 128), 256, 0, stream>>>(
        ctx, ao_w + (size_t)l * HH * HH, ao_b + (size_t)l * HH,
        delta, MM, HH, HH);
    ln_residual_kernel<<<MM, 256, 0, stream>>>(x, delta,
        ln1_s + (size_t)l * HH, ln1_b + (size_t)l * HH, x);
    gemm_bias_kernel<1><<<dim3(FFF / 64, MM / 128), 256, 0, stream>>>(
        x, ff1_w + (size_t)l * HH * FFF, ff1_b + (size_t)l * FFF,
        hmid, MM, FFF, HH);
    gemm_bias_kernel<0><<<dim3(HH / 64, MM / 128), 256, 0, stream>>>(
        hmid, ff2_w + (size_t)l * FFF * HH, ff2_b + (size_t)l * HH,
        delta, MM, HH, FFF);
    ln_residual_kernel<<<MM, 256, 0, stream>>>(x, delta,
        ln2_s + (size_t)l * HH, ln2_b + (size_t)l * HH, x);
  }

  head_kernel<<<MM, 256, 0, stream>>>(x, head_w, head_b, em);
  crf_forward_kernel<<<BB, 32, 0, stream>>>(em, labels, amask, trans,
                                            start_t, end_t, out);
  viterbi_kernel<<<BB, 32, 0, stream>>>(em, amask, trans, start_t, end_t,
                                        hist, out + 1);
}
